// TransformerLayer_29850022707650
// MI455X (gfx1250) — matrix operand order repaired and checked by reading, not compiled, not run on hardware
//
#include <hip/hip_runtime.h>
#include <hip/hip_bf16.h>
#include <math.h>

// ---------------------------------------------------------------------------
// Types for CDNA5 WMMA (wave32): A/B = 16 halves (8 VGPRs), C/D = 8 floats.
// ---------------------------------------------------------------------------
typedef __attribute__((ext_vector_type(16))) _Float16 v16h;
typedef __attribute__((ext_vector_type(8)))  _Float16 v8h;
typedef __attribute__((ext_vector_type(8)))  float    v8f;

__device__ __forceinline__ v8f wmma_f16(v16h a, v16h b, v8f c) {
    // D = A(16x32) * B(32x16) + C ; emits v_wmma_f32_16x16x32_f16
    return __builtin_amdgcn_wmma_f32_16x16x32_f16(
        /*neg_a=*/false, a, /*neg_b=*/false, b,
        /*c_mod=*/(short)0, c, /*reuse_a=*/false, /*reuse_b=*/false);
}

// Load 16 contiguous halves as a v16h via two 16B-aligned v8h loads.
__device__ __forceinline__ v16h ld16(const _Float16* p) {
    v8h lo = *(const v8h*)p;
    v8h hi = *(const v8h*)(p + 8);
    return __builtin_shufflevector(lo, hi, 0,1,2,3,4,5,6,7,8,9,10,11,12,13,14,15);
}

// A-fragment (16x32, f16) from row-major [*, lda] matrix.
// This code's contiguous k order (matches B frags): lane (m = l&15, hi = l>>4): elem i -> k = hi*16+i
__device__ __forceinline__ v16h load_a_frag(const _Float16* A, int row, int kb, int lda) {
    int lane = threadIdx.x & 31;
    int m = lane & 15, hi = lane >> 4;
    const _Float16* p = A + (size_t)(row + m) * lda + kb + hi * 16;
    v8h lo = *(const v8h*)p;
    v8h hh = *(const v8h*)(p + 8);
    return __builtin_shufflevector(lo, hh, 0,1,2,3,4,5,6,7,8,9,10,11,12,13,14,15);
}

// B-fragment (32x16) from TRANSPOSED B, i.e. Bt is [N][K] row-major.
// This code's contiguous k order (matches A frags): lane (n = l&15, hi = l>>4): b[i] = B[hi*16+i][n] = Bt[n][kb + hi*16 + i]
__device__ __forceinline__ v16h load_b_frag_t(const _Float16* Bt, int col, int kb, int ldb) {
    int lane = threadIdx.x & 31;
    int n = lane & 15, hi = lane >> 4;
    return ld16(Bt + (size_t)(col + n) * ldb + kb + hi * 16);
}

// ---------------------------------------------------------------------------
// Weight cast + transpose: in f32 [R][C]  ->  out f16 [C][R]
// ---------------------------------------------------------------------------
__global__ __launch_bounds__(256)
void transpose_cast_k(const float* __restrict__ in, _Float16* __restrict__ out,
                      int R, int C) {
    __shared__ float tile[32][33];
    int tx = threadIdx.x & 31;
    int ty = threadIdx.x >> 5;           // 0..7
    int cb = blockIdx.x * 32;
    int rb = blockIdx.y * 32;
    #pragma unroll
    for (int i = ty; i < 32; i += 8) {
        int r = rb + i, c = cb + tx;
        tile[i][tx] = (r < R && c < C) ? in[(size_t)r * C + c] : 0.f;
    }
    __syncthreads();
    #pragma unroll
    for (int i = ty; i < 32; i += 8) {
        int orow = cb + i;               // original col
        int ocol = rb + tx;              // original row
        if (orow < C && ocol < R)
            out[(size_t)orow * R + ocol] = (_Float16)tile[tx][i];
    }
}

// ---------------------------------------------------------------------------
// LayerNorm (fp32 in) -> f16 out. One block per row of D elements.
// ---------------------------------------------------------------------------
__global__ __launch_bounds__(256)
void layernorm_k(const float* __restrict__ x, const float* __restrict__ g,
                 const float* __restrict__ b, _Float16* __restrict__ out, int D) {
    __shared__ float sh[256], sh2[256];
    int row = blockIdx.x;
    int tid = threadIdx.x;
    const float* xr = x + (size_t)row * D;
    float s = 0.f, s2 = 0.f;
    for (int i = tid; i < D; i += 256) { float v = xr[i]; s += v; s2 += v * v; }
    sh[tid] = s; sh2[tid] = s2;
    __syncthreads();
    for (int off = 128; off > 0; off >>= 1) {
        if (tid < off) { sh[tid] += sh[tid + off]; sh2[tid] += sh2[tid + off]; }
        __syncthreads();
    }
    float mu   = sh[0] / (float)D;
    float var  = sh2[0] / (float)D - mu * mu;
    float rstd = rsqrtf(var + 1e-5f);
    _Float16* orow = out + (size_t)row * D;
    for (int i = tid; i < D; i += 256)
        orow[i] = (_Float16)((xr[i] - mu) * rstd * g[i] + b[i]);
}

// ---------------------------------------------------------------------------
// WMMA GEMM: C[M,N] = A[M,K] * B[K,N]  with Bt = B^T stored [N][K] f16.
// 4 waves per block; wave w owns a 32x64 strip: rows [blockIdx.x*128 + w*32),
// cols [blockIdx.y*64, +64) as 2 M-tiles x 4 N-subtiles (8 WMMA per k-step).
// K-loop unrolled by 2 with ping-pong fragment buffers: each prefetch loads
// directly into the register set the *next* half-step's WMMAs consume, so no
// v_mov copies appear in the hot loop and global_load_b128 clauses overlap
// the matrix ops. Requires Kd % 64 == 0 (true for all call sites: 1024/4096).
// Epilogue modes:
//   0: outH = C                       (q/k/v/sim projections)
//   2: outF = resid + C + bias        (attention out + residual)
//   3: outH = gelu(C + bias)          (FFN up)
//   4: outF = resid + C + bias        (FFN down -> final output)
// ---------------------------------------------------------------------------
__global__ __launch_bounds__(128)
void gemm_wmma_k(const _Float16* __restrict__ A, const _Float16* __restrict__ Bt,
                 float* __restrict__ outF, _Float16* __restrict__ outH,
                 const float* __restrict__ bias, const float* __restrict__ resid,
                 int M, int N, int Kd, int mode) {
    int wave = threadIdx.x >> 5;
    int lane = threadIdx.x & 31;
    int row_base = blockIdx.x * 128 + wave * 32;
    int col_base = blockIdx.y * 64;
    if (row_base >= M) return;

    v8f acc[2][4] = {};

    // prime: buffer 0 <- k-step 0
    v16h a00 = load_a_frag(A, row_base,      0, Kd);
    v16h a01 = load_a_frag(A, row_base + 16, 0, Kd);
    v16h b0[4];
    #pragma unroll
    for (int t = 0; t < 4; ++t) b0[t] = load_b_frag_t(Bt, col_base + t * 16, 0, Kd);

    for (int kb = 0; kb < Kd; kb += 64) {
        // prefetch buffer 1 <- kb+32 (always valid: Kd % 64 == 0)
        int k1 = kb + 32;
        v16h a10 = load_a_frag(A, row_base,      k1, Kd);
        v16h a11 = load_a_frag(A, row_base + 16, k1, Kd);
        v16h b1[4];
        #pragma unroll
        for (int t = 0; t < 4; ++t) b1[t] = load_b_frag_t(Bt, col_base + t * 16, k1, Kd);

        // compute with buffer 0 (kb)
        #pragma unroll
        for (int t = 0; t < 4; ++t) {
            acc[0][t] = wmma_f16(a00, b0[t], acc[0][t]);
            acc[1][t] = wmma_f16(a01, b0[t], acc[1][t]);
        }

        // prefetch buffer 0 <- kb+64 (wraps to 0 on the last trip: redundant
        // but unconditional, keeps the load clause formation clean)
        int k2 = (kb + 64 < Kd) ? (kb + 64) : 0;
        a00 = load_a_frag(A, row_base,      k2, Kd);
        a01 = load_a_frag(A, row_base + 16, k2, Kd);
        #pragma unroll
        for (int t = 0; t < 4; ++t) b0[t] = load_b_frag_t(Bt, col_base + t * 16, k2, Kd);

        // compute with buffer 1 (kb+32)
        #pragma unroll
        for (int t = 0; t < 4; ++t) {
            acc[0][t] = wmma_f16(a10, b1[t], acc[0][t]);
            acc[1][t] = wmma_f16(a11, b1[t], acc[1][t]);
        }
    }

    int n = lane & 15, hi = lane >> 4;
    #pragma unroll
    for (int mi = 0; mi < 2; ++mi) {
        #pragma unroll
        for (int t = 0; t < 4; ++t) {
            int col = col_base + t * 16 + n;
            float bs = (mode == 0) ? 0.f : bias[col];
            #pragma unroll
            for (int r = 0; r < 8; ++r) {
                int row = row_base + mi * 16 + r + 8 * hi;
                size_t idx = (size_t)row * N + col;
                float v = acc[mi][t][r] + bs;
                if (mode == 2 || mode == 4) {
                    outF[idx] = resid[idx] + v;
                } else if (mode == 3) {
                    outH[idx] = (_Float16)(0.5f * v * (1.f + erff(v * 0.70710678118654752f)));
                } else {
                    outH[idx] = (_Float16)v;
                }
            }
        }
    }
}

// ---------------------------------------------------------------------------
// Row-normalize the sim projection (f16, SIM_D=256 cols). One wave per row.
// ---------------------------------------------------------------------------
__global__ __launch_bounds__(256)
void sim_normalize_k(_Float16* __restrict__ sim, int cols) {
    int row  = blockIdx.x * 8 + (threadIdx.x >> 5);
    int lane = threadIdx.x & 31;
    _Float16* p = sim + (size_t)row * cols;
    float s = 0.f;
    for (int i = lane; i < cols; i += 32) { float v = (float)p[i]; s += v * v; }
    #pragma unroll
    for (int off = 16; off > 0; off >>= 1) s += __shfl_xor(s, off, 32);
    float sc = 1.f / fmaxf(sqrtf(s), 1e-12f);
    for (int i = lane; i < cols; i += 32) p[i] = (_Float16)((float)p[i] * sc);
}

// ---------------------------------------------------------------------------
// Flash attention with similarity bias.
//   scores = (Q K^T) * hd^-0.5 + 0.2 * (Qs Ks^T),  softmax over keys, * V.
// Grid: (B*H, Kseq/64).  Block: 128 threads (4 waves); wave owns 16 q rows.
// Key blocks of 32 staged in LDS; scale folded into Q frags, 0.2 into Qs frags
// so one WMMA accumulator chain produces the biased scores directly.
// ---------------------------------------------------------------------------
__global__ __launch_bounds__(128)
void attn_flash_k(const _Float16* __restrict__ q, const _Float16* __restrict__ k,
                  const _Float16* __restrict__ v, const _Float16* __restrict__ sim,
                  _Float16* __restrict__ ctx, int Kseq, int D, int simd) {
    const int HD = 64;
    __shared__ _Float16 Kt[32][72];        // key rows [key][d]
    __shared__ _Float16 Vt[64][40];        // transposed V [d][key]
    __shared__ _Float16 Ks[32][264];       // sim rows for keys
    __shared__ _Float16 Pst[4][16][40];    // per-wave P staging (C->A layout)

    int bh = blockIdx.x;
    int h  = bh & 15;
    int b  = bh >> 4;
    int tid  = threadIdx.x;
    int wave = tid >> 5;
    int lane = tid & 31;
    int m  = lane & 15;
    int hi = lane >> 4;
    int qrow0 = blockIdx.y * 64 + wave * 16;

    const _Float16* qbase = q   + (size_t)b * Kseq * D + h * HD;
    const _Float16* kbase = k   + (size_t)b * Kseq * D + h * HD;
    const _Float16* vbase = v   + (size_t)b * Kseq * D + h * HD;
    const _Float16* sbase = sim + (size_t)b * Kseq * simd;

    // Q fragments (hd=64 -> 2 k-steps), pre-scaled by hd^-0.5 = 0.125
    v16h qfrag[2];
    #pragma unroll
    for (int kk = 0; kk < 2; ++kk) {
        const _Float16* p = qbase + (size_t)(qrow0 + m) * D + kk * 32 + hi * 16;
        v16h a;
        #pragma unroll
        for (int i = 0; i < 8; ++i) a[i]     = (_Float16)((float)p[i]      * 0.125f);
        #pragma unroll
        for (int i = 0; i < 8; ++i) a[8 + i] = (_Float16)((float)p[8 + i] * 0.125f);
        qfrag[kk] = a;
    }
    // Q-sim fragments (256 -> 8 k-steps), pre-scaled by SIM_BOOST = 0.2
    v16h sfrag[8];
    #pragma unroll
    for (int kk = 0; kk < 8; ++kk) {
        const _Float16* p = sbase + (size_t)(qrow0 + m) * simd + kk * 32 + hi * 16;
        v16h a;
        #pragma unroll
        for (int i = 0; i < 8; ++i) a[i]     = (_Float16)((float)p[i]      * 0.2f);
        #pragma unroll
        for (int i = 0; i < 8; ++i) a[8 + i] = (_Float16)((float)p[8 + i] * 0.2f);
        sfrag[kk] = a;
    }

    v8f ctxacc[4] = {};
    float mrow[8], lrow[8];
    #pragma unroll
    for (int r = 0; r < 8; ++r) { mrow[r] = -1e30f; lrow[r] = 0.f; }

    for (int kb = 0; kb < Kseq; kb += 32) {
        // cooperative LDS staging (128 threads)
        for (int idx = tid; idx < 32 * 64; idx += 128) {
            int kk = idx >> 6, d = idx & 63;
            Kt[kk][d] = kbase[(size_t)(kb + kk) * D + d];
            Vt[d][kk] = vbase[(size_t)(kb + kk) * D + d];
        }
        for (int idx = tid; idx < 32 * 256; idx += 128) {
            int kk = idx >> 8, d = idx & 255;
            Ks[kk][d] = sbase[(size_t)(kb + kk) * simd + d];
        }
        __syncthreads();

        // scores tile: 16 q-rows x 32 keys (two 16-wide N subtiles)
        v8f s0 = {}, s1 = {};
        #pragma unroll
        for (int kk = 0; kk < 2; ++kk) {
            v16h b0 = ld16(&Kt[m][kk * 32 + hi * 16]);
            v16h b1 = ld16(&Kt[16 + m][kk * 32 + hi * 16]);
            s0 = wmma_f16(qfrag[kk], b0, s0);
            s1 = wmma_f16(qfrag[kk], b1, s1);
        }
        #pragma unroll
        for (int kk = 0; kk < 8; ++kk) {
            v16h b0 = ld16(&Ks[m][kk * 32 + hi * 16]);
            v16h b1 = ld16(&Ks[16 + m][kk * 32 + hi * 16]);
            s0 = wmma_f16(sfrag[kk], b0, s0);
            s1 = wmma_f16(sfrag[kk], b1, s1);
        }

        // online softmax: row r+8*hi lives in 16 lanes sharing `hi`
        float corr[8];
        #pragma unroll
        for (int r = 0; r < 8; ++r) {
            float vmax = fmaxf(s0[r], s1[r]);
            #pragma unroll
            for (int off = 1; off < 16; off <<= 1)
                vmax = fmaxf(vmax, __shfl_xor(vmax, off, 32));
            float newm = fmaxf(mrow[r], vmax);
            corr[r] = __expf(mrow[r] - newm);
            mrow[r] = newm;
            float e0 = __expf(s0[r] - newm);
            float e1 = __expf(s1[r] - newm);
            s0[r] = e0; s1[r] = e1;
            float rs = e0 + e1;
            #pragma unroll
            for (int off = 1; off < 16; off <<= 1)
                rs += __shfl_xor(rs, off, 32);
            lrow[r] = lrow[r] * corr[r] + rs;
        }
        #pragma unroll
        for (int t = 0; t < 4; ++t)
            #pragma unroll
            for (int r = 0; r < 8; ++r)
                ctxacc[t][r] *= corr[r];

        // C-layout P -> A-layout via per-wave LDS staging
        #pragma unroll
        for (int r = 0; r < 8; ++r) {
            Pst[wave][r + 8 * hi][m]      = (_Float16)s0[r];
            Pst[wave][r + 8 * hi][16 + m] = (_Float16)s1[r];
        }
        v16h pf;
        {
            const _Float16* pp = &Pst[wave][m][hi * 16];
            v8h lo = *(const v8h*)pp;
            v8h hh = *(const v8h*)(pp + 8);
            pf = __builtin_shufflevector(lo, hh, 0,1,2,3,4,5,6,7,8,9,10,11,12,13,14,15);
        }
        // ctx += P(16x32) * V(32x64)
        #pragma unroll
        for (int t = 0; t < 4; ++t) {
            v16h vf = ld16(&Vt[t * 16 + m][hi * 16]);
            ctxacc[t] = wmma_f16(pf, vf, ctxacc[t]);
        }
        __syncthreads();
    }

    // finalize: divide by row sums, write f16 ctx [B*Kseq, D]
    _Float16* obase = ctx + ((size_t)b * Kseq + qrow0) * D + h * HD;
    #pragma unroll
    for (int t = 0; t < 4; ++t) {
        int col = t * 16 + m;
        #pragma unroll
        for (int r = 0; r < 8; ++r) {
            int row = r + 8 * hi;
            obase[(size_t)row * D + col] = (_Float16)(ctxacc[t][r] / lrow[r]);
        }
    }
}

// ---------------------------------------------------------------------------
// Host-side orchestration
// ---------------------------------------------------------------------------
extern "C" void kernel_launch(void* const* d_in, const int* in_sizes, int n_in,
                              void* d_out, int out_size, void* d_ws, size_t ws_size,
                              hipStream_t stream) {
    (void)in_sizes; (void)n_in; (void)out_size; (void)ws_size;
    const int B = 2, Kseq = 2048, D = 1024, H = 16, FF = 4096, SD = 256;
    const int M = B * Kseq;   // 4096 token rows

    const float* x    = (const float*)d_in[0];
    const float* Wq   = (const float*)d_in[1];
    const float* Wk   = (const float*)d_in[2];
    const float* Wv   = (const float*)d_in[3];
    const float* Wo   = (const float*)d_in[4];
    const float* bo   = (const float*)d_in[5];
    const float* Wsim = (const float*)d_in[6];
    const float* g1   = (const float*)d_in[7];
    const float* be1  = (const float*)d_in[8];
    const float* g2   = (const float*)d_in[9];
    const float* be2  = (const float*)d_in[10];
    const float* W1   = (const float*)d_in[11];
    const float* b1   = (const float*)d_in[12];
    const float* W2   = (const float*)d_in[13];
    const float* b2   = (const float*)d_in[14];
    float* out = (float*)d_out;

    // workspace carve-up (~123 MB total), 256B-aligned regions
    char* base = (char*)d_ws;
    size_t off = 0;
    auto allocH = [&](size_t n) -> _Float16* {
        _Float16* p = (_Float16*)(base + off);
        off += ((n * sizeof(_Float16)) + 255) & ~(size_t)255;
        return p;
    };
    auto allocF = [&](size_t n) -> float* {
        float* p = (float*)(base + off);
        off += ((n * sizeof(float)) + 255) & ~(size_t)255;
        return p;
    };
    _Float16* WqT   = allocH((size_t)D * D);
    _Float16* WkT   = allocH((size_t)D * D);
    _Float16* WvT   = allocH((size_t)D * D);
    _Float16* WoT   = allocH((size_t)D * D);
    _Float16* WsimT = allocH((size_t)SD * D);
    _Float16* W1T   = allocH((size_t)FF * D);
    _Float16* W2T   = allocH((size_t)D * FF);
    _Float16* nx    = allocH((size_t)M * D);
    _Float16* qh    = allocH((size_t)M * D);
    _Float16* kh    = allocH((size_t)M * D);
    _Float16* vh    = allocH((size_t)M * D);
    _Float16* simh  = allocH((size_t)M * SD);
    _Float16* ctxh  = allocH((size_t)M * D);
    _Float16* nx2   = allocH((size_t)M * D);
    _Float16* hh    = allocH((size_t)M * FF);
    float*    xres  = allocF((size_t)M * D);

    // 1) weights -> f16 transposed [N][K]
    transpose_cast_k<<<dim3(D / 32, D / 32), 256, 0, stream>>>(Wq,   WqT,   D,  D);
    transpose_cast_k<<<dim3(D / 32, D / 32), 256, 0, stream>>>(Wk,   WkT,   D,  D);
    transpose_cast_k<<<dim3(D / 32, D / 32), 256, 0, stream>>>(Wv,   WvT,   D,  D);
    transpose_cast_k<<<dim3(D / 32, D / 32), 256, 0, stream>>>(Wo,   WoT,   D,  D);
    transpose_cast_k<<<dim3(SD / 32, D / 32), 256, 0, stream>>>(Wsim, WsimT, D,  SD);
    transpose_cast_k<<<dim3(FF / 32, D / 32), 256, 0, stream>>>(W1,   W1T,   D,  FF);
    transpose_cast_k<<<dim3(D / 32, FF / 32), 256, 0, stream>>>(W2,   W2T,   FF, D);

    // 2) LN1
    layernorm_k<<<M, 256, 0, stream>>>(x, g1, be1, nx, D);

    // 3) q/k/v/sim projections (mode 0), block tile = 128 rows x 64 cols
    gemm_wmma_k<<<dim3(M / 128, D / 64), 128, 0, stream>>>(nx, WqT, nullptr, qh, nullptr, nullptr, M, D, D, 0);
    gemm_wmma_k<<<dim3(M / 128, D / 64), 128, 0, stream>>>(nx, WkT, nullptr, kh, nullptr, nullptr, M, D, D, 0);
    gemm_wmma_k<<<dim3(M / 128, D / 64), 128, 0, stream>>>(nx, WvT, nullptr, vh, nullptr, nullptr, M, D, D, 0);
    gemm_wmma_k<<<dim3(M / 128, SD / 64), 128, 0, stream>>>(nx, WsimT, nullptr, simh, nullptr, nullptr, M, SD, D, 0);

    // 4) normalize sim rows
    sim_normalize_k<<<M / 8, 256, 0, stream>>>(simh, SD);

    // 5) flash attention with sim bias
    attn_flash_k<<<dim3(B * H, Kseq / 64), 128, 0, stream>>>(qh, kh, vh, simh, ctxh, Kseq, D, SD);

    // 6) attn out projection + residual: xres = x + ctx@Wo + bo (mode 2)
    gemm_wmma_k<<<dim3(M / 128, D / 64), 128, 0, stream>>>(ctxh, WoT, xres, nullptr, bo, x, M, D, D, 2);

    // 7) LN2
    layernorm_k<<<M, 256, 0, stream>>>(xres, g2, be2, nx2, D);

    // 8) FFN up with exact GELU (mode 3)
    gemm_wmma_k<<<dim3(M / 128, FF / 64), 128, 0, stream>>>(nx2, W1T, nullptr, hh, b1, nullptr, M, FF, D, 3);

    // 9) FFN down + residual -> final output (mode 4)
    gemm_wmma_k<<<dim3(M / 128, D / 64), 128, 0, stream>>>(hh, W2T, out, nullptr, b2, xres, M, D, FF, 4);
}
